// RelativeMultiHeadAttention_6700148981874
// MI455X (gfx1250) — matrix operand order repaired and checked by reading, not compiled, not run on hardware
//
#include <hip/hip_runtime.h>

#define B_    2
#define TQ_   1024
#define TMEM_ 1024
#define TK_   2048
#define IN_   1024
#define HD_   1024
#define H_    16
#define D_    64
#define OUT_  1024

typedef __attribute__((ext_vector_type(16))) __bf16 v16bf;
typedef __attribute__((ext_vector_type(8)))  float  v8f;

__device__ __forceinline__ unsigned short f2bf(float f) {
    unsigned int u = __float_as_uint(f);
    u = (u + 0x7fffu + ((u >> 16) & 1u)) >> 16;   // round-to-nearest-even
    return (unsigned short)u;
}

union BFU { v16bf v; unsigned int u[8]; };

// A operand fragment (16x32 bf16). lane = row (l%16); half (l/16) selects
// K subset: VGPR j<4 -> K = 8*half + 2j ; j>=4 -> K = 16 + 8*half + 2(j-4).
__device__ __forceinline__ v16bf load_fragA(const unsigned short* base, int hl) {
    BFU f;
#pragma unroll
    for (int j = 0; j < 8; ++j) {
        int koff = ((j < 4) ? (2 * j) : (16 + 2 * (j - 4))) + 8 * hl;
        f.u[j] = *(const unsigned int*)(base + koff);
    }
    return f.v;
}

// B operand fragment (32x16 bf16), sourced as [N][K] rows. lane = column
// (l%16); VGPR j<4 -> K = 8*half + 2j ; j>=4 -> K = 16 + 8*half + 2(j-4).
__device__ __forceinline__ v16bf load_fragB(const unsigned short* base, int hl) {
    BFU f;
#pragma unroll
    for (int j = 0; j < 8; ++j) {
        int koff = ((j < 4) ? (2 * j) : (16 + 2 * (j - 4))) + 8 * hl;
        f.u[j] = *(const unsigned int*)(base + koff);
    }
    return f.v;
}

__device__ __forceinline__ v8f wmma_bf16(v16bf a, v16bf b, v8f c) {
    return __builtin_amdgcn_wmma_f32_16x16x32_bf16(false, a, false, b,
                                                   (short)0, c, false, false);
}

// ---------------- conversion kernels ----------------

__global__ void k_f32_to_bf16(const float* __restrict__ src,
                              unsigned short* __restrict__ dst, long n) {
    long i = (long)blockIdx.x * blockDim.x + threadIdx.x;
    if (i < n) dst[i] = f2bf(src[i]);
}

// full = concat(memory, x) along time, converted to bf16. [B, TK, IN]
__global__ void k_build_full(const float* __restrict__ mem,
                             const float* __restrict__ x,
                             unsigned short* __restrict__ full) {
    long i = (long)blockIdx.x * blockDim.x + threadIdx.x;
    long n = (long)B_ * TK_ * IN_;
    if (i >= n) return;
    long b = i / ((long)TK_ * IN_);
    long r = i - b * (long)TK_ * IN_;
    long t = r / IN_;
    long c = r - t * IN_;
    float v = (t < TMEM_) ? mem[(b * TMEM_ + t) * IN_ + c]
                          : x[(b * TQ_ + (t - TMEM_)) * IN_ + c];
    full[i] = f2bf(v);
}

// WT[n*K + k] = W[k*N + n], fp32 -> bf16
__global__ void k_transpose_bf16(const float* __restrict__ W,
                                 unsigned short* __restrict__ WT,
                                 int K, int N) {
    long i = (long)blockIdx.x * blockDim.x + threadIdx.x;
    if (i >= (long)K * N) return;
    int n = (int)(i / K);
    int k = (int)(i - (long)n * K);
    WT[i] = f2bf(W[(long)k * N + n]);
}

// ---------------- generic NT bf16 GEMM ----------------
// C[z] = A[z] (MxK) * Bt[z]^T (Bt stored [N][K]) (+ bias[col])
// block: 256 threads (8 waves), tile 128x128, wave tile 32x64 (2x4 WMMA accs)
// Templated on bias/output format so the epilogue is branch-free.
template <bool HAS_BIAS, bool C_BF16>
__global__ __launch_bounds__(256) void k_gemm_nt(
    const unsigned short* __restrict__ A, const unsigned short* __restrict__ Bt,
    void* __restrict__ Cv, int Kd, int lda, int ldb, int ldc,
    long long sAb, long long sAh, long long sBb, long long sBh,
    long long sCb, long long sCh, int zH,
    const float* __restrict__ bias) {
    int tid = threadIdx.x;
    int w = tid >> 5, l = tid & 31, hl = l >> 4, ln = l & 15;
    int m0 = blockIdx.y * 128 + (w & 3) * 32;
    int n0 = blockIdx.x * 128 + (w >> 2) * 64;
    long long zb = blockIdx.z / zH, zh = blockIdx.z % zH;
    const unsigned short* Ab = A + zb * sAb + zh * sAh;
    const unsigned short* Bb = Bt + zb * sBb + zh * sBh;

    v8f acc[2][4];
#pragma unroll
    for (int mt = 0; mt < 2; ++mt)
#pragma unroll
        for (int nt = 0; nt < 4; ++nt)
            acc[mt][nt] = (v8f){0.f, 0.f, 0.f, 0.f, 0.f, 0.f, 0.f, 0.f};

    for (int kb = 0; kb < Kd; kb += 32) {
        v16bf aF[2], bF[4];
#pragma unroll
        for (int mt = 0; mt < 2; ++mt)
            aF[mt] = load_fragA(Ab + (long long)(m0 + mt * 16 + ln) * lda + kb, hl);
#pragma unroll
        for (int nt = 0; nt < 4; ++nt)
            bF[nt] = load_fragB(Bb + (long long)(n0 + nt * 16 + ln) * ldb + kb, hl);
#pragma unroll
        for (int mt = 0; mt < 2; ++mt)
#pragma unroll
            for (int nt = 0; nt < 4; ++nt)
                acc[mt][nt] = wmma_bf16(aF[mt], bF[nt], acc[mt][nt]);
    }

    long long cbase = zb * sCb + zh * sCh;
#pragma unroll
    for (int mt = 0; mt < 2; ++mt) {
#pragma unroll
        for (int nt = 0; nt < 4; ++nt) {
            int col = n0 + nt * 16 + ln;
            float bv = HAS_BIAS ? bias[col] : 0.f;
            long long coff = cbase + (long long)(m0 + mt * 16 + 8 * hl) * ldc + col;
#pragma unroll
            for (int r = 0; r < 8; ++r) {
                float val = acc[mt][nt][r] + bv;
                if (C_BF16) ((unsigned short*)Cv)[coff] = f2bf(val);
                else        ((float*)Cv)[coff] = val;
                coff += ldc;
            }
        }
    }
}

// ---------------- flash attention with rel-shifted pos scores ----------------
// grid: (TQ/128, B*H). 8 waves, each owns a 16-query strip; online softmax
// over 32 key-blocks of 64. S = Qc*K^T (WMMA, NT direct from global/L2),
// pos gathered from precomputed PS with the rel_shift index map, O = P*V
// (P re-laid-out via per-wave LDS, V transposed in shared LDS).
__global__ __launch_bounds__(256) void k_attn(
    const unsigned short* __restrict__ Qc, const unsigned short* __restrict__ Kb,
    const unsigned short* __restrict__ Vb, const float* __restrict__ PS,
    const unsigned char* __restrict__ mask, unsigned short* __restrict__ Obf) {
    __shared__ unsigned short lds_p[8 * 16 * 64];  // per-wave P tiles
    __shared__ unsigned short lds_v[64 * 64];      // V^T tile [d][k]

    int tid = threadIdx.x;
    int w = tid >> 5, l = tid & 31, hl = l >> 4, ln = l & 15;
    int bh = blockIdx.y;
    int b = bh >> 4, h = bh & (H_ - 1);
    int qfr = blockIdx.x * 128 + w * 16 + ln;       // A-fragment row (query)
    int q0 = blockIdx.x * 128 + w * 16 + 8 * hl;    // first stat row (r=0)

    const unsigned short* Qrow = Qc + ((long long)b * TQ_ + qfr) * HD_ + h * D_;
    v16bf aQ0 = load_fragA(Qrow, hl);
    v16bf aQ1 = load_fragA(Qrow + 32, hl);

    const float* PSbh = PS + (long long)bh * TQ_ * TQ_;
    const unsigned char* mrow_base = mask + (long long)b * TQ_ * TK_ + (long long)q0 * TK_;

    float mrow[8], lrow[8];
    v8f acc[4];
#pragma unroll
    for (int r = 0; r < 8; ++r) { mrow[r] = -3.0e38f; lrow[r] = 0.f; }
#pragma unroll
    for (int dt = 0; dt < 4; ++dt)
        acc[dt] = (v8f){0.f, 0.f, 0.f, 0.f, 0.f, 0.f, 0.f, 0.f};

    for (int kb = 0; kb < TK_ / 64; ++kb) {
        int kgb = kb * 64;

        // ---- S = Qc * K^T for this 128x64 block (wave strip: 16x64) ----
        v8f S[4];
#pragma unroll
        for (int nt = 0; nt < 4; ++nt) {
            int krow = kgb + nt * 16 + ln;
            const unsigned short* Kr = Kb + ((long long)b * TK_ + krow) * HD_ + h * D_;
            v16bf b0 = load_fragB(Kr, hl);
            v16bf b1 = load_fragB(Kr + 32, hl);
            v8f z = (v8f){0.f, 0.f, 0.f, 0.f, 0.f, 0.f, 0.f, 0.f};
            S[nt] = wmma_bf16(aQ1, b1, wmma_bf16(aQ0, b0, z));
        }

        // ---- rel-shifted pos add, scale, mask ----
#pragma unroll
        for (int r = 0; r < 8; ++r) {
            int q = q0 + r;
            const unsigned char* mr = mrow_base + (long long)r * TK_ + kgb;
            const float* psq  = PSbh + (long long)q * TQ_;
            const float* psq1 = psq + TQ_;
#pragma unroll
            for (int nt = 0; nt < 4; ++nt) {
                int kg = kgb + nt * 16 + ln;
                float s = S[nt][r];
                if (kg >= TMEM_) {
                    int j = kg - TMEM_;
                    float pv = 0.f;
                    if (j < q)      pv = psq[TQ_ - q + j];
                    else if (j > q) pv = psq1[j - q - 1];
                    s += pv;
                }
                s *= 0.125f;  // 1/sqrt(64)
                if (!mr[nt * 16 + ln]) s = -3.4e38f;
                S[nt][r] = s;
            }
        }

        // ---- online softmax update (row stats across 16 lanes of a half) ----
#pragma unroll
        for (int r = 0; r < 8; ++r) {
            float vmax = fmaxf(fmaxf(S[0][r], S[1][r]), fmaxf(S[2][r], S[3][r]));
#pragma unroll
            for (int off = 1; off < 16; off <<= 1)
                vmax = fmaxf(vmax, __shfl_xor(vmax, off, 32));
            float mnew = fmaxf(mrow[r], vmax);
            float sc = __expf(mrow[r] - mnew);
            mrow[r] = mnew;
            float rs = 0.f;
#pragma unroll
            for (int nt = 0; nt < 4; ++nt) {
                float p = __expf(S[nt][r] - mnew);
                S[nt][r] = p;
                rs += p;
            }
#pragma unroll
            for (int off = 1; off < 16; off <<= 1)
                rs += __shfl_xor(rs, off, 32);
            lrow[r] = lrow[r] * sc + rs;
#pragma unroll
            for (int dt = 0; dt < 4; ++dt) acc[dt][r] *= sc;
        }

        // ---- P (C-layout) -> per-wave LDS [16][64] bf16 ----
#pragma unroll
        for (int nt = 0; nt < 4; ++nt)
#pragma unroll
            for (int r = 0; r < 8; ++r)
                lds_p[w * 1024 + (8 * hl + r) * 64 + nt * 16 + ln] = f2bf(S[nt][r]);

        __syncthreads();  // prior iteration's lds_v reads are done
        // ---- stage V^T [d][k] cooperatively ----
#pragma unroll
        for (int i = 0; i < 16; ++i) {
            int idx = tid + i * 256;
            int kp = idx >> 6, d = idx & 63;
            lds_v[d * 64 + kp] =
                Vb[((long long)b * TK_ + kgb + kp) * HD_ + h * D_ + d];
        }
        __syncthreads();

        // ---- acc += P * V ----
        const unsigned short* pb = &lds_p[w * 1024 + ln * 64];
        v16bf aP0 = load_fragA(pb, hl);
        v16bf aP1 = load_fragA(pb + 32, hl);
#pragma unroll
        for (int dt = 0; dt < 4; ++dt) {
            const unsigned short* vb = &lds_v[(dt * 16 + ln) * 64];
            v16bf b0 = load_fragB(vb, hl);
            v16bf b1 = load_fragB(vb + 32, hl);
            acc[dt] = wmma_bf16(aP1, b1, wmma_bf16(aP0, b0, acc[dt]));
        }
    }

    // ---- normalize and emit O (bf16, [B,TQ,H*D]) ----
#pragma unroll
    for (int dt = 0; dt < 4; ++dt)
#pragma unroll
        for (int r = 0; r < 8; ++r) {
            int q = q0 + r;
            float o = acc[dt][r] / lrow[r];
            Obf[((long long)b * TQ_ + q) * HD_ + h * D_ + dt * 16 + ln] = f2bf(o);
        }
}

// ---------------- host orchestration ----------------

extern "C" void kernel_launch(void* const* d_in, const int* in_sizes, int n_in,
                              void* d_out, int out_size, void* d_ws, size_t ws_size,
                              hipStream_t stream) {
    (void)in_sizes; (void)n_in; (void)out_size; (void)ws_size;
    const float* x      = (const float*)d_in[0];
    const float* memory = (const float*)d_in[1];
    const unsigned char* mask = (const unsigned char*)d_in[2];
    const float* Wq   = (const float*)d_in[3];
    const float* Wk   = (const float*)d_in[4];
    const float* Wv   = (const float*)d_in[5];
    const float* Wpos = (const float*)d_in[6];
    const float* u    = (const float*)d_in[7];
    const float* v2   = (const float*)d_in[8];
    const float* Wout = (const float*)d_in[9];
    float* out = (float*)d_out;

    char* ws = (char*)d_ws;
    size_t off = 0;
    auto alloc = [&](size_t bytes) -> void* {
        void* p = ws + off;
        off = (off + bytes + 255) & ~(size_t)255;
        return p;
    };
    unsigned short* xbf    = (unsigned short*)alloc((size_t)B_ * TQ_ * IN_ * 2);
    unsigned short* fullbf = (unsigned short*)alloc((size_t)B_ * TK_ * IN_ * 2);
    unsigned short* WqT    = (unsigned short*)alloc((size_t)IN_ * HD_ * 2);
    unsigned short* WkT    = (unsigned short*)alloc((size_t)IN_ * HD_ * 2);
    unsigned short* WvT    = (unsigned short*)alloc((size_t)IN_ * HD_ * 2);
    unsigned short* WposT  = (unsigned short*)alloc((size_t)IN_ * HD_ * 2);
    unsigned short* WoutT  = (unsigned short*)alloc((size_t)HD_ * OUT_ * 2);
    unsigned short* Qc     = (unsigned short*)alloc((size_t)B_ * TQ_ * HD_ * 2);
    unsigned short* Qp     = (unsigned short*)alloc((size_t)B_ * TQ_ * HD_ * 2);
    unsigned short* Kbf    = (unsigned short*)alloc((size_t)B_ * TK_ * HD_ * 2);
    unsigned short* Vbf    = (unsigned short*)alloc((size_t)B_ * TK_ * HD_ * 2);
    unsigned short* Pbf    = (unsigned short*)alloc((size_t)B_ * TQ_ * HD_ * 2);
    unsigned short* Obf    = (unsigned short*)alloc((size_t)B_ * TQ_ * HD_ * 2);
    float* PS = (float*)alloc((size_t)B_ * H_ * TQ_ * TQ_ * 4);

    // conversions
    long nx = (long)B_ * TQ_ * IN_;
    k_f32_to_bf16<<<dim3((unsigned)((nx + 255) / 256)), 256, 0, stream>>>(x, xbf, nx);
    long nf = (long)B_ * TK_ * IN_;
    k_build_full<<<dim3((unsigned)((nf + 255) / 256)), 256, 0, stream>>>(memory, x, fullbf);
    long nw = (long)IN_ * HD_;
    unsigned wb = (unsigned)((nw + 255) / 256);
    k_transpose_bf16<<<wb, 256, 0, stream>>>(Wq, WqT, IN_, HD_);
    k_transpose_bf16<<<wb, 256, 0, stream>>>(Wk, WkT, IN_, HD_);
    k_transpose_bf16<<<wb, 256, 0, stream>>>(Wv, WvT, IN_, HD_);
    k_transpose_bf16<<<wb, 256, 0, stream>>>(Wpos, WposT, IN_, HD_);
    k_transpose_bf16<<<wb, 256, 0, stream>>>(Wout, WoutT, HD_, OUT_);

    // projections (M folded over batch; x/full are contiguous [B*T, IN])
    {
        dim3 g(HD_ / 128, (B_ * TQ_) / 128, 1);
        k_gemm_nt<true, true><<<g, 256, 0, stream>>>(
            xbf, WqT, Qc, IN_, IN_, IN_, HD_, 0, 0, 0, 0, 0, 0, 1, u);
        k_gemm_nt<true, true><<<g, 256, 0, stream>>>(
            xbf, WqT, Qp, IN_, IN_, IN_, HD_, 0, 0, 0, 0, 0, 0, 1, v2);
        k_gemm_nt<false, true><<<g, 256, 0, stream>>>(
            xbf, WposT, Pbf, IN_, IN_, IN_, HD_, 0, 0, 0, 0, 0, 0, 1, nullptr);
    }
    {
        dim3 g(HD_ / 128, (B_ * TK_) / 128, 1);
        k_gemm_nt<false, true><<<g, 256, 0, stream>>>(
            fullbf, WkT, Kbf, IN_, IN_, IN_, HD_, 0, 0, 0, 0, 0, 0, 1, nullptr);
        k_gemm_nt<false, true><<<g, 256, 0, stream>>>(
            fullbf, WvT, Vbf, IN_, IN_, IN_, HD_, 0, 0, 0, 0, 0, 0, 1, nullptr);
    }

    // PS[b,h] = Qp_h (TQxD) * Pb_h^T  -> f32 [B,H,TQ,TQ]
    {
        dim3 g(TQ_ / 128, TQ_ / 128, B_ * H_);
        k_gemm_nt<false, false><<<g, 256, 0, stream>>>(
            Qp, Pbf, PS, D_, HD_, HD_, TQ_,
            (long long)TQ_ * HD_, D_, (long long)TQ_ * HD_, D_,
            (long long)H_ * TQ_ * TQ_, (long long)TQ_ * TQ_, H_, nullptr);
    }

    // fused flash attention
    k_attn<<<dim3(TQ_ / 128, B_ * H_), 256, 0, stream>>>(Qc, Kbf, Vbf, PS, mask, Obf);

    // final projection -> fp32 output
    {
        dim3 g(OUT_ / 128, (B_ * TQ_) / 128, 1);
        k_gemm_nt<false, false><<<g, 256, 0, stream>>>(
            Obf, WoutT, out, HD_, HD_, HD_, OUT_, 0, 0, 0, 0, 0, 0, 1, nullptr);
    }
}
